// GraphAttentionLayer_33002528703087
// MI455X (gfx1250) — matrix operand order repaired and checked by reading, not compiled, not run on hardware
//
#include <hip/hip_runtime.h>
#include <hip/hip_bf16.h>
#include <stddef.h>

// ---------------------------------------------------------------------------
// GAT layer, fused for MI455X (gfx1250, wave32, WMMA).
//   N = 8192, IN = 512, OUT = 64
//   Pass A: Wh = h @ W (fp32, exact), plus bf16 transposed copy WhT[64][8192]
//   Pass B: Wh1 = Wh@a1, Wh2 = Wh@a2 (one wave per row)
//   Pass C: fused attention: per (16-row tile, 32-j chunk)
//             E[m][k] = adj ? exp(leaky(Wh1[m]*Wh2[k])) : 0   (f32 VALU+trans)
//             num    += E_bf16 (A) x WhT_bf16 (B)   via v_wmma_f32_16x16x32_bf16
//             s[m]   += sum_k E (pure f32)
//           partials combined with global f32 atomics.
//   Pass D: out = elu(num / s)
// ---------------------------------------------------------------------------

typedef __attribute__((ext_vector_type(16))) __bf16 v16bf;
typedef __attribute__((ext_vector_type(8)))  float  v8f;
typedef __attribute__((ext_vector_type(4)))  float  f32x4;
typedef __attribute__((ext_vector_type(4)))  int    i32x4;

#define GAT_N      8192
#define GAT_IN     512
#define GAT_OUT    64
#define GAT_ALPHA  0.2f
#define GAT_LOG2E  1.44269504088896340736f
#define GAT_PARTS  8          // j-partitions per row tile (occupancy)

// ws layout (bytes)
#define WS_NUM   ((size_t)0)                                   // 8192*64 f32
#define WS_S     (WS_NUM + (size_t)GAT_N * GAT_OUT * 4)        // 8192 f32
#define WS_WH    (WS_S   + (size_t)GAT_N * 4)                  // 8192*64 f32
#define WS_WHT   (WS_WH  + (size_t)GAT_N * GAT_OUT * 4)        // 64*8192 bf16
#define WS_WH1   (WS_WHT + (size_t)GAT_OUT * GAT_N * 2)        // 8192 f32
#define WS_WH2   (WS_WH1 + (size_t)GAT_N * 4)                  // 8192 f32

// ---------------------------------------------------------------------------
// Pass A: Wh = h @ W   (block = 256 thr = 8 waves, 16 rows per block)
// thread (c = t&63, rg = t>>6) computes rows rg*4 .. rg*4+3, column c.
// ---------------------------------------------------------------------------
__global__ void __launch_bounds__(256) gat_wh_kernel(
    const float* __restrict__ h, const float* __restrict__ W,
    float* __restrict__ Wh, __bf16* __restrict__ WhT)
{
  __shared__ float sh[16 * GAT_IN];          // 32 KB
  const int i0 = blockIdx.x * 16;

  for (int idx = threadIdx.x; idx < 16 * GAT_IN; idx += 256) {
    int r = idx >> 9, c = idx & (GAT_IN - 1);
    sh[idx] = h[(size_t)(i0 + r) * GAT_IN + c];
  }
  __syncthreads();

  const int c  = threadIdx.x & 63;
  const int rg = threadIdx.x >> 6;           // uniform per wave
  float acc[4] = {0.f, 0.f, 0.f, 0.f};

  for (int k4 = 0; k4 < GAT_IN / 4; ++k4) {
    const int k = k4 * 4;
    float w0 = W[(k + 0) * GAT_OUT + c];
    float w1 = W[(k + 1) * GAT_OUT + c];
    float w2 = W[(k + 2) * GAT_OUT + c];
    float w3 = W[(k + 3) * GAT_OUT + c];
#pragma unroll
    for (int q = 0; q < 4; ++q) {
      f32x4 hv = *(const f32x4*)&sh[(rg * 4 + q) * GAT_IN + k];
      acc[q] += hv[0] * w0 + hv[1] * w1 + hv[2] * w2 + hv[3] * w3;
    }
  }

#pragma unroll
  for (int q = 0; q < 4; ++q) {
    int row = i0 + rg * 4 + q;
    Wh[(size_t)row * GAT_OUT + c] = acc[q];
    WhT[(size_t)c * GAT_N + row] = (__bf16)acc[q];
  }
}

// ---------------------------------------------------------------------------
// Pass B: Wh1/Wh2 — one wave (32 lanes) per row, 2 cols per lane + shfl reduce
// ---------------------------------------------------------------------------
__global__ void __launch_bounds__(256) gat_wh12_kernel(
    const float* __restrict__ Wh, const float* __restrict__ a1,
    const float* __restrict__ a2, float* __restrict__ Wh1,
    float* __restrict__ Wh2)
{
  const int row = blockIdx.x * 8 + (threadIdx.x >> 5);
  const int l   = threadIdx.x & 31;
  const float* wr = Wh + (size_t)row * GAT_OUT;
  float x0 = wr[l], x1 = wr[l + 32];
  float v1 = x0 * a1[l] + x1 * a1[l + 32];
  float v2 = x0 * a2[l] + x1 * a2[l + 32];
#pragma unroll
  for (int off = 16; off > 0; off >>= 1) {
    v1 += __shfl_xor(v1, off, 32);
    v2 += __shfl_xor(v2, off, 32);
  }
  if (l == 0) { Wh1[row] = v1; Wh2[row] = v2; }
}

// ---------------------------------------------------------------------------
// Pass C: fused attention + WMMA accumulation.
// Wave w handles (tile = w/PARTS, part = w%PARTS): rows [tile*16, +16),
// j in [part*1024, +1024) step 32.  A-matrix lane layout (16-bit, 16x32):
//   lane L holds row m = L&15, K(e) = (L>>4)*16 + e, e=0..15
// B-matrix (bf16 32x16): lane L holds col n = L&15, K = (L>>4)*16 + e
// D (f32 16x16): vgpr r -> row (L>>4)*8 + r, col L&15
// ---------------------------------------------------------------------------
__global__ void __launch_bounds__(256) gat_att_kernel(
    const int* __restrict__ adj,
    const float* __restrict__ Wh1, const float* __restrict__ Wh2,
    const __bf16* __restrict__ WhT,
    float* __restrict__ num, float* __restrict__ sden)
{
  const int wid  = blockIdx.x * 8 + (threadIdx.x >> 5);
  const int tile = wid / GAT_PARTS;
  const int part = wid % GAT_PARTS;
  const int i0   = tile * 16;
  const int lane = threadIdx.x & 31;
  const int m    = lane & 15;
  const int half = lane >> 4;
  const int row  = i0 + m;

  const float wh1 = Wh1[row];
  const int jbeg = part * (GAT_N / GAT_PARTS);
  const int jend = jbeg + (GAT_N / GAT_PARTS);

  const int* __restrict__ adjrow = adj + (size_t)row * GAT_N;

  v8f acc[4];
#pragma unroll
  for (int ct = 0; ct < 4; ++ct) acc[ct] = (v8f)0.f;
  float spart = 0.f;

  for (int j0 = jbeg; j0 < jend; j0 += 32) {
    // prefetch next adj chunk for this row (speculative, RT)
    __builtin_prefetch(adjrow + j0 + 32, 0, 1);

    // Wh2 gathers in A layout: one run of 16 consecutive floats
    const float* w2p = Wh2 + j0 + half * 16;
    f32x4 wa = *(const f32x4*)(w2p);
    f32x4 wb = *(const f32x4*)(w2p + 4);
    f32x4 wc = *(const f32x4*)(w2p + 8);
    f32x4 wd = *(const f32x4*)(w2p + 12);

    const int* ap = adjrow + j0 + half * 16;
    i32x4 ma = *(const i32x4*)(ap);
    i32x4 mb = *(const i32x4*)(ap + 4);
    i32x4 mc = *(const i32x4*)(ap + 8);
    i32x4 md = *(const i32x4*)(ap + 12);

    float w2v[16];
    int   mv[16];
#pragma unroll
    for (int t = 0; t < 4; ++t) {
      w2v[t] = wa[t];      mv[t] = ma[t];
      w2v[4 + t] = wb[t];  mv[4 + t] = mb[t];
      w2v[8 + t] = wc[t];  mv[8 + t] = mc[t];
      w2v[12 + t] = wd[t]; mv[12 + t] = md[t];
    }

    v16bf A;
    float sp = 0.f;
#pragma unroll
    for (int e = 0; e < 16; ++e) {
      float att = wh1 * w2v[e];
      float lk  = fmaxf(att, GAT_ALPHA * att);          // leaky relu
      float ex  = __builtin_amdgcn_exp2f(lk * GAT_LOG2E);
      ex = (mv[e] > 0) ? ex : 0.f;                      // adjacency mask
      sp += ex;
      A[e] = (__bf16)ex;
    }
    spart += sp;

#pragma unroll
    for (int ct = 0; ct < 4; ++ct) {
      const v16bf* bp = (const v16bf*)(WhT + (size_t)(ct * 16 + m) * GAT_N +
                                       j0 + half * 16);
      v16bf B = *bp;
      acc[ct] = __builtin_amdgcn_wmma_f32_16x16x32_bf16(
          false, A, false, B, (short)0, acc[ct], false, false);
    }
  }

  // denominator: row m covered by lanes m (K low halves) and m+16 (K high)
  float stot = spart + __shfl_xor(spart, 16, 32);
  if (half == 0) unsafeAtomicAdd(sden + row, stot);

  // numerator partials: D layout — vgpr r -> row half*8 + r, col ct*16 + m
#pragma unroll
  for (int ct = 0; ct < 4; ++ct) {
#pragma unroll
    for (int r = 0; r < 8; ++r) {
      unsafeAtomicAdd(num + (size_t)(i0 + half * 8 + r) * GAT_OUT + ct * 16 + m,
                      acc[ct][r]);
    }
  }
}

// ---------------------------------------------------------------------------
// Pass D: out = elu(num / s)
// ---------------------------------------------------------------------------
__global__ void __launch_bounds__(256) gat_final_kernel(
    const float* __restrict__ num, const float* __restrict__ sden,
    float* __restrict__ out)
{
  const int idx = blockIdx.x * 256 + threadIdx.x;   // N*OUT threads
  float x = num[idx] / sden[idx >> 6];
  out[idx] = (x > 0.f) ? x : (__builtin_amdgcn_exp2f(x * GAT_LOG2E) - 1.f);
}

// ---------------------------------------------------------------------------
extern "C" void kernel_launch(void* const* d_in, const int* in_sizes, int n_in,
                              void* d_out, int out_size, void* d_ws,
                              size_t ws_size, hipStream_t stream) {
  const float* h   = (const float*)d_in[0];   // [8192,512] f32
  const int*   adj = (const int*)  d_in[1];   // [8192,8192] i32
  const float* W   = (const float*)d_in[2];   // [512,64] f32
  const float* a1  = (const float*)d_in[3];   // [64] f32
  const float* a2  = (const float*)d_in[4];   // [64] f32
  float* out = (float*)d_out;                 // [8192,64] f32

  char* ws = (char*)d_ws;
  float*  num  = (float*) (ws + WS_NUM);
  float*  sden = (float*) (ws + WS_S);
  float*  Wh   = (float*) (ws + WS_WH);
  __bf16* WhT  = (__bf16*)(ws + WS_WHT);
  float*  Wh1  = (float*) (ws + WS_WH1);
  float*  Wh2  = (float*) (ws + WS_WH2);

  // zero the atomic accumulators (num + sden) every launch (capture-safe)
  hipMemsetAsync(d_ws, 0, (size_t)(GAT_N * GAT_OUT + GAT_N) * sizeof(float),
                 stream);

  gat_wh_kernel<<<GAT_N / 16, 256, 0, stream>>>(h, W, Wh, WhT);
  gat_wh12_kernel<<<GAT_N / 8, 256, 0, stream>>>(Wh, a1, a2, Wh1, Wh2);
  gat_att_kernel<<<(GAT_N / 16) * GAT_PARTS / 8, 256, 0, stream>>>(
      adj, Wh1, Wh2, WhT, num, sden);
  gat_final_kernel<<<(GAT_N * GAT_OUT) / 256, 256, 0, stream>>>(num, sden, out);
}
